// TimeMixing_7962869367246
// MI455X (gfx1250) — compile-verified, matrix operand order checked by reading, not run on hardware
//
#include <hip/hip_runtime.h>
#include <hip/hip_bf16.h>

// ---------------------------------------------------------------------------
// RWKV time-mixing for MI455X (gfx1250, wave32).
//   - bf16 WMMA (v_wmma_f32_16x16x32_bf16) for the four 32768x1024x1024 GEMMs
//   - LDS-tiled 128x128 block GEMM, 8 waves x (32x64) per wave
//   - Tensor Data Mover (tensor_load_to_lds) stages bf16 tiles global->LDS
//     with D# LDS padding for bank-conflict-free ds_load_b128 fragment reads
//   - fp32 scalar scan for the sequential wkv recurrence
// ---------------------------------------------------------------------------

typedef __attribute__((ext_vector_type(16))) __bf16 v16bf;
typedef __attribute__((ext_vector_type(8)))  __bf16 v8bf;
typedef __attribute__((ext_vector_type(8)))  float  v8f;
typedef __attribute__((ext_vector_type(4)))  float  v4f;
typedef __attribute__((ext_vector_type(4)))  unsigned int v4u;
typedef __attribute__((ext_vector_type(8)))  int    v8i;
typedef __attribute__((ext_vector_type(4)))  int    v4i;

#define LDS_PITCH 40   // bf16 elements per LDS row: 32 data + 8 pad (80 B)

static __device__ __forceinline__ __bf16 to_bf16(float f) {
  unsigned int u = __builtin_bit_cast(unsigned int, f);
  unsigned int r = u + 0x7FFFu + ((u >> 16) & 1u);   // round to nearest even
  unsigned short h = (unsigned short)(r >> 16);
  return __builtin_bit_cast(__bf16, h);
}

static __device__ __forceinline__ v8f wmma_bf16(v16bf a, v16bf b, v8f c) {
  return __builtin_amdgcn_wmma_f32_16x16x32_bf16(false, a, false, b,
                                                 (short)0, c, false, false);
}

// ---------------------------------------------------------------------------
// TDM: DMA a 2D tile (tile_rows x tile_cols bf16, row stride = stride_elems)
// from global memory into LDS at lds_byte_off, inserting 4 DWORDs of padding
// after every 16 DWORDs (64B row -> 80B LDS pitch).  D# per CDNA5 ISA ch.8.
// ---------------------------------------------------------------------------
static __device__ __forceinline__ void
tdm_load_tile_2d(const void* gptr, unsigned lds_byte_off,
                 unsigned tile_cols, unsigned tile_rows,
                 unsigned tensor_d0, unsigned tensor_d1,
                 unsigned long long stride_elems) {
  unsigned long long ga = (unsigned long long)(size_t)gptr;
  v4u g0;
  g0[0] = 1u;                                        // count=1 (valid), user
  g0[1] = lds_byte_off;                              // lds_addr
  g0[2] = (unsigned)(ga & 0xFFFFFFFFu);              // global_addr[31:0]
  g0[3] = (unsigned)((ga >> 32) & 0x01FFFFFFu)       // global_addr[56:32]
        | (2u << 30);                                // type=2 (image)
  v8i g1;
  g1[0] = (int)((1u << 16)          // data_size = 2 bytes
              | (1u << 20)          // pad_enable
              | (3u << 22)          // pad_interval: 16 DWORDs (64 B)
              | (3u << 25));        // pad_amount:   4 DWORDs (16 B)
  g1[1] = (int)((tensor_d0 & 0xFFFFu) << 16);                         // dim0 lo
  g1[2] = (int)(((tensor_d0 >> 16) & 0xFFFFu) | ((tensor_d1 & 0xFFFFu) << 16));
  g1[3] = (int)(((tensor_d1 >> 16) & 0xFFFFu) | ((tile_cols & 0xFFFFu) << 16));
  g1[4] = (int)(tile_rows & 0xFFFFu);                // tile_dim1; tile_dim2=0
  g1[5] = (int)(stride_elems & 0xFFFFFFFFu);         // dim0_stride[31:0]
  g1[6] = (int)((stride_elems >> 32) & 0xFFFFu);     // dim0_stride[47:32]
  g1[7] = 0;
  v4i z4 = {0, 0, 0, 0};
#if __clang_major__ >= 23
  v8i z8 = {0, 0, 0, 0, 0, 0, 0, 0};
  __builtin_amdgcn_tensor_load_to_lds(g0, g1, z4, z4, z8, 0);
#else
  __builtin_amdgcn_tensor_load_to_lds(g0, g1, z4, z4, 0);
#endif
}

// ---------------------------------------------------------------------------
// Fragment load from LDS (padded pitch).  CDNA5 16-bit A/B layout:
//   lane L<16 : row rowbase+L, K 0..7  and 16..23
//   lane L+16 : row rowbase+L, K 8..15 and 24..31
// Two ds_load_b128 per fragment, concat with shufflevector (no b16 moves).
// ---------------------------------------------------------------------------
static __device__ __forceinline__ v16bf
load_frag_lds(const __bf16* lds, int rowbase, int lane) {
  int row  = rowbase + (lane & 15);
  int koff = (lane >> 4) * 8;
  const __bf16* p = lds + row * LDS_PITCH + koff;
  v8bf c0 = *(const v8bf*)(p);
  v8bf c1 = *(const v8bf*)(p + 16);
  return __builtin_shufflevector(c0, c1, 0, 1, 2, 3, 4, 5, 6, 7,
                                 8, 9, 10, 11, 12, 13, 14, 15);
}

// C/D 16x16 f32: VGPR i -> (M=i, N=lane) lanes 0-15; (M=8+i, N=lane-16) else.
static __device__ __forceinline__ void
store_frag_c(float* out, int ld, int mbase, int nbase, int lane,
             v8f c, const float* bias) {
  int n  = nbase + (lane & 15);
  int m0 = mbase + (lane >> 4) * 8;
  float bv = bias ? bias[n] : 0.0f;
#pragma unroll
  for (int i = 0; i < 8; ++i)
    out[(size_t)(m0 + i) * ld + n] = c[i] + bv;
}

// ---------------------------------------------------------------------------
// Kernel 1: weights fp32 -> bf16 (4 matrices packed contiguously, row-major)
// ---------------------------------------------------------------------------
__global__ void convert_weights_kernel(const float* __restrict__ Wk,
                                       const float* __restrict__ Wv,
                                       const float* __restrict__ Wr,
                                       const float* __restrict__ Wo,
                                       __bf16* __restrict__ dst, int dd) {
  int i = blockIdx.x * 256 + threadIdx.x;
  int which = i / dd;
  int off   = i - which * dd;
  const float* src = (which == 0) ? Wk : (which == 1) ? Wv
                   : (which == 2) ? Wr : Wo;
  dst[i] = to_bf16(src[off]);
}

// ---------------------------------------------------------------------------
// Kernel 2: fused time-shift mix (+sigmoid for r) + GEMM vs W^T.
// blockIdx.z: 0->k, 1->v, 2->r.  Block tile 128x128; A staged via VALU mix,
// B (weights) staged via TDM.  8 waves, each 32(M) x 64(N).
// ---------------------------------------------------------------------------
__global__ void __launch_bounds__(256)
gemm_mix_kernel(const float* __restrict__ x,
                const float* __restrict__ tmk,
                const float* __restrict__ tmv,
                const float* __restrict__ tmr,
                const __bf16* __restrict__ Wb,
                float* __restrict__ kout,
                float* __restrict__ vout,
                float* __restrict__ rout,
                int d, int T) {
  __shared__ __align__(16) __bf16 ldsA[128 * LDS_PITCH];
  __shared__ __align__(16) __bf16 ldsB[128 * LDS_PITCH];

  const int lane = threadIdx.x & 31;
  const int wave = threadIdx.x >> 5;
  const int wm = wave >> 1, wn = wave & 1;
  const int blockM = blockIdx.x * 128;
  const int blockN = blockIdx.y * 128;
  const int z = blockIdx.z;

  const float* tm = (z == 0) ? tmk : (z == 1) ? tmv : tmr;
  const __bf16* W = Wb + (size_t)z * d * d;
  float* out      = (z == 0) ? kout : (z == 1) ? vout : rout;
  const int sig = (z == 2);

  // staging role for this thread: one (row, 16-col half) of the 128x32 A tile
  const int arow = threadIdx.x >> 1;
  const int acol = (threadIdx.x & 1) * 16;
  const int grow = blockM + arow;
  const bool hasPrev = (grow % T) != 0;
  const float* xrow = x + (size_t)grow * d;

  v8f acc[2][4];
#pragma unroll
  for (int i = 0; i < 2; ++i)
#pragma unroll
    for (int j = 0; j < 4; ++j) acc[i][j] = (v8f){};

  for (int kb = 0; kb < d; kb += 32) {
    // ---- stage A (mix, fp32->bf16) into padded LDS -----------------------
    {
      const float* cur = xrow + kb + acol;
      v4f c[4], p[4], m[4];
#pragma unroll
      for (int q = 0; q < 4; ++q) {
        c[q] = *(const v4f*)(cur + q * 4);
        m[q] = *(const v4f*)(tm + kb + acol + q * 4);
        p[q] = (v4f){0.f, 0.f, 0.f, 0.f};
      }
      if (hasPrev) {
#pragma unroll
        for (int q = 0; q < 4; ++q) p[q] = *(const v4f*)(cur - d + q * 4);
      }
      v8bf o0, o1;
#pragma unroll
      for (int q = 0; q < 2; ++q)
#pragma unroll
        for (int j = 0; j < 4; ++j) {
          float v1 = c[q][j] * m[q][j] + p[q][j] * (1.0f - m[q][j]);
          if (sig) v1 = 1.0f / (1.0f + __expf(-v1));
          o0[q * 4 + j] = to_bf16(v1);
        }
#pragma unroll
      for (int q = 2; q < 4; ++q)
#pragma unroll
        for (int j = 0; j < 4; ++j) {
          float v1 = c[q][j] * m[q][j] + p[q][j] * (1.0f - m[q][j]);
          if (sig) v1 = 1.0f / (1.0f + __expf(-v1));
          o1[(q - 2) * 4 + j] = to_bf16(v1);
        }
      *(v8bf*)(&ldsA[arow * LDS_PITCH + acol])     = o0;
      *(v8bf*)(&ldsA[arow * LDS_PITCH + acol + 8]) = o1;
    }
    // ---- stage B (128 rows of W, 32 cols) via TDM ------------------------
    if (wave == 0) {
      tdm_load_tile_2d(W + (size_t)blockN * d + kb,
                       (unsigned)(size_t)(&ldsB[0]),
                       /*tile_cols=*/32, /*tile_rows=*/128,
                       /*tensor_d0=*/(unsigned)d, /*tensor_d1=*/(unsigned)d,
                       /*stride=*/(unsigned long long)d);
      __builtin_amdgcn_s_wait_tensorcnt(0);
    }
    __syncthreads();

    // ---- compute ---------------------------------------------------------
    v16bf af[2], bf[4];
#pragma unroll
    for (int i = 0; i < 2; ++i) af[i] = load_frag_lds(ldsA, wm * 32 + i * 16, lane);
#pragma unroll
    for (int j = 0; j < 4; ++j) bf[j] = load_frag_lds(ldsB, wn * 64 + j * 16, lane);
#pragma unroll
    for (int i = 0; i < 2; ++i)
#pragma unroll
      for (int j = 0; j < 4; ++j) acc[i][j] = wmma_bf16(af[i], bf[j], acc[i][j]);
    __syncthreads();
  }

#pragma unroll
  for (int i = 0; i < 2; ++i)
#pragma unroll
    for (int j = 0; j < 4; ++j)
      store_frag_c(out, d, blockM + wm * 32 + i * 16,
                   blockN + wn * 64 + j * 16, lane, acc[i][j], nullptr);
}

// ---------------------------------------------------------------------------
// Kernel 3: sequential wkv scan, one thread per (batch, channel).
// Writes s = wkv * r as bf16 row-major (A matrix of the output GEMM).
// ---------------------------------------------------------------------------
__global__ void wkv_scan_kernel(const float* __restrict__ k,
                                const float* __restrict__ v,
                                const float* __restrict__ r,
                                const float* __restrict__ u,
                                const float* __restrict__ w,
                                __bf16* __restrict__ s,
                                int T, int d) {
  int c  = blockIdx.x * blockDim.x + threadIdx.x;   // [0, B*d)
  int b  = c / d;
  int ch = c - b * d;
  float uc = u[ch], wc = w[ch];
  float a = 0.0f, bb = 0.0f;
  size_t idx = (size_t)b * T * d + ch;
  for (int t = 0; t < T; ++t, idx += d) {
    float kt = k[idx], vt = v[idx];
    float q  = fmaxf(uc + kt, wc);
    float e1 = __expf(-wc - q);
    float e2 = __expf(uc + kt - q);
    a  = e1 * a  + e2 * vt;
    bb = e1 * bb + e2;
    s[idx] = to_bf16((a / bb) * r[idx]);
  }
}

// ---------------------------------------------------------------------------
// Kernel 4: out = s @ Wo^T + bo.  Both A and B tiles staged via TDM.
// ---------------------------------------------------------------------------
__global__ void __launch_bounds__(256)
gemm_out_kernel(const __bf16* __restrict__ s,
                const __bf16* __restrict__ Wob,
                const float* __restrict__ bo,
                float* __restrict__ out, int d, int BT) {
  __shared__ __align__(16) __bf16 ldsA[128 * LDS_PITCH];
  __shared__ __align__(16) __bf16 ldsB[128 * LDS_PITCH];

  const int lane = threadIdx.x & 31;
  const int wave = threadIdx.x >> 5;
  const int wm = wave >> 1, wn = wave & 1;
  const int blockM = blockIdx.x * 128;
  const int blockN = blockIdx.y * 128;

  v8f acc[2][4];
#pragma unroll
  for (int i = 0; i < 2; ++i)
#pragma unroll
    for (int j = 0; j < 4; ++j) acc[i][j] = (v8f){};

  for (int kb = 0; kb < d; kb += 32) {
    if (wave == 0) {
      tdm_load_tile_2d(s + (size_t)blockM * d + kb,
                       (unsigned)(size_t)(&ldsA[0]),
                       32, 128, (unsigned)d, (unsigned)BT,
                       (unsigned long long)d);
      tdm_load_tile_2d(Wob + (size_t)blockN * d + kb,
                       (unsigned)(size_t)(&ldsB[0]),
                       32, 128, (unsigned)d, (unsigned)d,
                       (unsigned long long)d);
      __builtin_amdgcn_s_wait_tensorcnt(0);
    }
    __syncthreads();

    v16bf af[2], bf[4];
#pragma unroll
    for (int i = 0; i < 2; ++i) af[i] = load_frag_lds(ldsA, wm * 32 + i * 16, lane);
#pragma unroll
    for (int j = 0; j < 4; ++j) bf[j] = load_frag_lds(ldsB, wn * 64 + j * 16, lane);
#pragma unroll
    for (int i = 0; i < 2; ++i)
#pragma unroll
      for (int j = 0; j < 4; ++j) acc[i][j] = wmma_bf16(af[i], bf[j], acc[i][j]);
    __syncthreads();
  }

#pragma unroll
  for (int i = 0; i < 2; ++i)
#pragma unroll
    for (int j = 0; j < 4; ++j)
      store_frag_c(out, d, blockM + wm * 32 + i * 16,
                   blockN + wn * 64 + j * 16, lane, acc[i][j], bo);
}

// ---------------------------------------------------------------------------
// Launcher
// ---------------------------------------------------------------------------
extern "C" void kernel_launch(void* const* d_in, const int* in_sizes, int n_in,
                              void* d_out, int out_size, void* d_ws, size_t ws_size,
                              hipStream_t stream) {
  const int d = 1024, B = 8, T = 4096;
  const int BT = B * T;

  const float* x   = (const float*)d_in[0];
  const float* u   = (const float*)d_in[1];
  const float* w   = (const float*)d_in[2];
  const float* tmk = (const float*)d_in[3];
  const float* tmv = (const float*)d_in[4];
  const float* tmr = (const float*)d_in[5];
  const float* Wk  = (const float*)d_in[6];
  const float* Wv  = (const float*)d_in[7];
  const float* Wr  = (const float*)d_in[8];
  const float* Wo  = (const float*)d_in[9];
  const float* bo  = (const float*)d_in[10];
  float* out = (float*)d_out;

  const size_t dd  = (size_t)d * d;
  const size_t btd = (size_t)BT * d;

  // Workspace layout
  char* ws = (char*)d_ws;
  __bf16* Wb   = (__bf16*)ws;                          // 4*dd bf16 = 8 MB
  float*  kbuf = (float*)(ws + 4 * dd * sizeof(__bf16));
  float*  vbuf = kbuf + btd;
  float*  rbuf = vbuf + btd;
  __bf16* sbuf = (__bf16*)(rbuf + btd);                // btd bf16

  convert_weights_kernel<<<dim3((unsigned)((4 * dd + 255) / 256)), 256, 0, stream>>>(
      Wk, Wv, Wr, Wo, Wb, (int)dd);

  dim3 g1(BT / 128, d / 128, 3);
  gemm_mix_kernel<<<g1, 256, 0, stream>>>(x, tmk, tmv, tmr, Wb,
                                          kbuf, vbuf, rbuf, d, T);

  wkv_scan_kernel<<<dim3(B * d / 256), 256, 0, stream>>>(kbuf, vbuf, rbuf,
                                                         u, w, sbuf, T, d);

  dim3 g2(BT / 128, d / 128);
  gemm_out_kernel<<<g2, 256, 0, stream>>>(sbuf, Wb + 3 * dd, bo, out, d, BT);
}
